// PointNetSetAbstraction_9698036154799
// MI455X (gfx1250) — compile-verified, matrix operand order checked by reading, not run on hardware
//
#include <hip/hip_runtime.h>
#include <hip/hip_bf16.h>

// ---------------------------------------------------------------------------
// PointNet++ Set Abstraction for MI455X (gfx1250, wave32, WMMA)
// B=16, N=4096, S=1024, K=32, D_FEAT=64, MLP=[64,64,128], IN_CH=67
// M = B*S*K = 524288 grouped points.
// MLP layers run as f16 WMMA GEMMs (v_wmma_f32_16x16x32_f16), f32 accumulate,
// with BatchNorm stats fused into the GEMM via wave shuffles + f32 atomics.
// ---------------------------------------------------------------------------

#define BQ   16
#define NQ   4096
#define SQ   1024
#define KQ   32
#define DF   64
#define CIN  67
#define C0P  96              // IN_CH padded to multiple of 32
#define MTOT (BQ * SQ * KQ)  // 524288

typedef __attribute__((ext_vector_type(16))) _Float16 v16h;
typedef __attribute__((ext_vector_type(8)))  _Float16 h8;
typedef __attribute__((ext_vector_type(8)))  float    v8f;

// ---- workspace layout (bytes) ---------------------------------------------
#define WS_CENT   ((size_t)0)                      // int   [B*S]
#define WS_NEWXYZ ((size_t)65536)                  // float [B*S][3]
#define WS_STATS  ((size_t)262144)                 // float [3][4][128] sum/sumsq/mu/rstd
#define WS_W0P    ((size_t)270336)                 // f16 [64][96]
#define WS_W1P    (WS_W0P + 64*96*2)               // f16 [64][64]
#define WS_W2P    (WS_W1P + 64*64*2)               // f16 [128][64]
#define WS_X0     ((size_t)1048576)                // f16 [M][96]
#define WS_Y      (WS_X0 + (size_t)MTOT*96*2)      // f16 [128][M]
#define WS_X1     (WS_Y  + (size_t)128*MTOT*2)     // f16 [M][64]
#define WS_X2     (WS_X1 + (size_t)MTOT*64*2)      // f16 [M][64]
// total ~371 MB

// ---------------------------------------------------------------------------
// Kernel 0: pack weights to padded f16, zero BN stats accumulators
// ---------------------------------------------------------------------------
__global__ void pack_weights_kernel(const float* __restrict__ w0,
                                    const float* __restrict__ w1,
                                    const float* __restrict__ w2,
                                    _Float16* __restrict__ w0p,
                                    _Float16* __restrict__ w1p,
                                    _Float16* __restrict__ w2p,
                                    float* __restrict__ stats) {
  for (int i = blockIdx.x * blockDim.x + threadIdx.x; i < 8192;
       i += gridDim.x * blockDim.x) {
    if (i < 1536) stats[i] = 0.0f;                       // 3 layers * 512
    if (i < 6144) {                                      // 64x96 (pad 67->96)
      int o = i / C0P, c = i % C0P;
      w0p[i] = (c < CIN) ? (_Float16)w0[o * CIN + c] : (_Float16)0.0f;
    }
    if (i < 4096) w1p[i] = (_Float16)w1[i];              // 64x64
    if (i < 8192) w2p[i] = (_Float16)w2[i];              // 128x64
  }
}

// ---------------------------------------------------------------------------
// Kernel 1: farthest point sampling (one workgroup per batch, cloud in LDS)
// ---------------------------------------------------------------------------
__global__ void fps_kernel(const float* __restrict__ xyz,   // [B,3,N]
                           const int* __restrict__ fps_start,
                           int* __restrict__ cent_idx,      // [B*S]
                           float* __restrict__ new_xyz_ws,  // [B*S][3]
                           float* __restrict__ out_newxyz)  // d_out [B,3,S]
{
  __shared__ float sx[NQ], sy[NQ], sz[NQ];
  __shared__ float rv[256];
  __shared__ int   ri[256];
  __shared__ int   scent[SQ];

  const int b = blockIdx.x, tid = threadIdx.x;
  for (int n = tid; n < NQ; n += 256) {
    sx[n] = xyz[(b * 3 + 0) * NQ + n];
    sy[n] = xyz[(b * 3 + 1) * NQ + n];
    sz[n] = xyz[(b * 3 + 2) * NQ + n];
  }
  float d[16];
#pragma unroll
  for (int j = 0; j < 16; ++j) d[j] = 1e10f;
  __syncthreads();

  int far = fps_start[b];
  for (int s = 0; s < SQ; ++s) {
    if (tid == 0) scent[s] = far;
    const float cx = sx[far], cy = sy[far], cz = sz[far];
    float bestv = -1.0f; int besti = 0;
#pragma unroll
    for (int j = 0; j < 16; ++j) {
      const int n = tid + 256 * j;
      const float dx = sx[n] - cx, dy = sy[n] - cy, dz = sz[n] - cz;
      const float dd = dx * dx + dy * dy + dz * dz;
      d[j] = fminf(d[j], dd);
      if (d[j] > bestv) { bestv = d[j]; besti = n; }
    }
    rv[tid] = bestv; ri[tid] = besti;
    __syncthreads();
    for (int off = 128; off > 0; off >>= 1) {
      if (tid < off) {
        const float ov = rv[tid + off]; const int oi = ri[tid + off];
        if (ov > rv[tid] || (ov == rv[tid] && oi < ri[tid])) {
          rv[tid] = ov; ri[tid] = oi;
        }
      }
      __syncthreads();
    }
    far = ri[0];
    __syncthreads();
  }
  for (int s = tid; s < SQ; s += 256) {
    const int c = scent[s];
    cent_idx[b * SQ + s] = c;
    const float X = sx[c], Y = sy[c], Z = sz[c];
    new_xyz_ws[(b * SQ + s) * 3 + 0] = X;
    new_xyz_ws[(b * SQ + s) * 3 + 1] = Y;
    new_xyz_ws[(b * SQ + s) * 3 + 2] = Z;
    out_newxyz[(b * 3 + 0) * SQ + s] = X;
    out_newxyz[(b * 3 + 1) * SQ + s] = Y;
    out_newxyz[(b * 3 + 2) * SQ + s] = Z;
  }
}

// ---------------------------------------------------------------------------
// Kernel 2: ball query (top-32 by distance, masked vs radius^2 quirk) +
//           gather + f16 pack into X0 [M][96] (WMMA B-fragment layout)
// ---------------------------------------------------------------------------
__global__ void ballquery_gather_kernel(const float* __restrict__ xyz,     // [B,3,N]
                                        const float* __restrict__ points,  // [B,64,N]
                                        const float* __restrict__ new_xyz, // [B*S][3]
                                        _Float16* __restrict__ X0)         // [M][96]
{
  __shared__ float dist[NQ];
  __shared__ float rv[256];
  __shared__ int   ri[256];
  __shared__ int   sel[KQ];
  __shared__ float selv[KQ];

  const int q = blockIdx.x;          // b*S + s
  const int b = q >> 10;
  const int tid = threadIdx.x;
  const float qx = new_xyz[q * 3 + 0];
  const float qy = new_xyz[q * 3 + 1];
  const float qz = new_xyz[q * 3 + 2];

  for (int n = tid; n < NQ; n += 256) {
    const float dx = xyz[(b * 3 + 0) * NQ + n] - qx;
    const float dy = xyz[(b * 3 + 1) * NQ + n] - qy;
    const float dz = xyz[(b * 3 + 2) * NQ + n] - qz;
    dist[n] = dx * dx + dy * dy + dz * dz;
  }
  __syncthreads();

  for (int k = 0; k < KQ; ++k) {
    float bestv = 3.4e38f; int besti = 0;
#pragma unroll
    for (int j = 0; j < 16; ++j) {
      const int n = tid + 256 * j;
      const float dd = dist[n];
      if (dd < bestv) { bestv = dd; besti = n; }
    }
    rv[tid] = bestv; ri[tid] = besti;
    __syncthreads();
    for (int off = 128; off > 0; off >>= 1) {
      if (tid < off) {
        const float ov = rv[tid + off]; const int oi = ri[tid + off];
        if (ov < rv[tid] || (ov == rv[tid] && oi < ri[tid])) {
          rv[tid] = ov; ri[tid] = oi;
        }
      }
      __syncthreads();
    }
    if (tid == 0) { sel[k] = ri[0]; selv[k] = rv[0]; dist[ri[0]] = 3.4e38f; }
    __syncthreads();
  }

  // mask: sqrt(sq) < radius^2 (0.04)  <=>  sq < 0.0016 (faithful to reference)
  const int i0 = sel[0];
  for (int t = tid; t < KQ * C0P; t += 256) {
    const int k = t / C0P, c = t - (t / C0P) * C0P;
    const int n = (selv[k] < 0.0016f) ? sel[k] : i0;
    float v;
    if (c < 3)        v = xyz[(b * 3 + c) * NQ + n] - new_xyz[q * 3 + c];
    else if (c < CIN) v = points[(b * DF + (c - 3)) * NQ + n];
    else              v = 0.0f;
    X0[(size_t)(q * KQ + k) * C0P + c] = (_Float16)v;
  }
}

// ---------------------------------------------------------------------------
// WMMA fragment loader: lane holds K = [8g, 8g+8) U [16+8g, 16+8g+8)
// within the current 32-wide K chunk (two aligned 16B loads).
// ---------------------------------------------------------------------------
static __device__ __forceinline__ v16h load_frag(const _Float16* p, int g) {
  const h8 lo = *(const h8*)(p + 8 * g);
  const h8 hi = *(const h8*)(p + 16 + 8 * g);
  return __builtin_shufflevector(lo, hi, 0, 1, 2, 3, 4, 5, 6, 7,
                                 8, 9, 10, 11, 12, 13, 14, 15);
}

// ---------------------------------------------------------------------------
// Kernel 3: GEMM  Y[Co][M] = W[Co][Ci] * X[Ci stored as [M][CiPad]]
// f16 in, f32 accumulate via v_wmma_f32_16x16x32_f16; fused BN sum/sumsq.
// grid = (M/128, Co/16), block = 256 (8 waves, one 16x16 tile each).
// ---------------------------------------------------------------------------
__global__ void gemm_wmma_kernel(const _Float16* __restrict__ X,
                                 const _Float16* __restrict__ W,
                                 _Float16* __restrict__ Y,
                                 float* __restrict__ stats,   // [0:128)=sum [128:256)=sumsq
                                 int M, int CiPad, int nkc) {
  const int lane = threadIdx.x & 31;
  const int wave = threadIdx.x >> 5;
  const int m_base  = (blockIdx.x * 8 + wave) * 16;
  const int co_base = blockIdx.y * 16;
  const int col = lane & 15;
  const int g   = lane >> 4;

  const _Float16* pa = W + (size_t)(co_base + col) * CiPad;
  const _Float16* pb = X + (size_t)(m_base + col) * CiPad;

  v8f acc = {};
  for (int kc = 0; kc < nkc; ++kc) {
    const v16h a = load_frag(pa + kc * 32, g);
    const v16h bf = load_frag(pb + kc * 32, g);
    acc = __builtin_amdgcn_wmma_f32_16x16x32_f16(
        false, a, false, bf, (short)0, acc, false, false);
  }

  const int m = m_base + col;
#pragma unroll
  for (int r = 0; r < 8; ++r) {
    const int co = co_base + r + 8 * g;
    const float v = acc[r];
    Y[(size_t)co * M + m] = (_Float16)v;
    float s1 = v, s2 = v * v;
#pragma unroll
    for (int off = 1; off < 16; off <<= 1) {
      s1 += __shfl_xor(s1, off, 32);
      s2 += __shfl_xor(s2, off, 32);
    }
    if (col == 0) {
      atomicAdd(&stats[co], s1);
      atomicAdd(&stats[128 + co], s2);
    }
  }
}

// ---------------------------------------------------------------------------
// Kernel 4: finalize BN stats: mu, rstd
// ---------------------------------------------------------------------------
__global__ void finalize_stats_kernel(float* __restrict__ stats, float invM) {
  const int c = threadIdx.x;
  const float mu  = stats[c] * invM;
  const float var = stats[128 + c] * invM - mu * mu;
  stats[256 + c] = mu;
  stats[384 + c] = rsqrtf(var + 1e-5f);
}

// ---------------------------------------------------------------------------
// Kernel 5: normalize + ReLU + repack f16 into next layer's [M][64] layout
// ---------------------------------------------------------------------------
__global__ void norm_relu_pack_kernel(const _Float16* __restrict__ Y,
                                      const float* __restrict__ stats,
                                      const float* __restrict__ gmm,
                                      const float* __restrict__ bet,
                                      _Float16* __restrict__ Xn, int M) {
  const unsigned t = blockIdx.x * 256u + threadIdx.x;   // over 64*M
  const int co = t / (unsigned)M;
  const unsigned m = t - (unsigned)co * (unsigned)M;
  const float y = (float)Y[t];
  float v = (y - stats[256 + co]) * stats[384 + co] * gmm[co] + bet[co];
  v = fmaxf(v, 0.0f);
  Xn[(size_t)m * 64 + co] = (_Float16)v;
}

// ---------------------------------------------------------------------------
// Kernel 6: layer-3 normalize + ReLU fused with max over K, write f32 output
// ---------------------------------------------------------------------------
__global__ void norm_relu_max_kernel(const _Float16* __restrict__ Y,
                                     const float* __restrict__ stats,
                                     const float* __restrict__ gmm,
                                     const float* __restrict__ bet,
                                     float* __restrict__ out, int M) {
  const unsigned t = blockIdx.x * 256u + threadIdx.x;   // over 128 * B*S
  const int co = t >> 14;
  const int bs = t & 16383;
  const _Float16* p = Y + (size_t)co * M + (size_t)bs * KQ;
  const float mu = stats[256 + co], rs = stats[384 + co];
  const float gg = gmm[co], bb = bet[co];
  float mx = 0.0f;
#pragma unroll
  for (int k = 0; k < KQ; ++k) {
    const float v = fmaxf(((float)p[k] - mu) * rs * gg + bb, 0.0f);
    mx = fmaxf(mx, v);
  }
  const int b = bs >> 10, s = bs & 1023;
  out[BQ * 3 * SQ + ((size_t)b * 128 + co) * SQ + s] = mx;
}

// ---------------------------------------------------------------------------
extern "C" void kernel_launch(void* const* d_in, const int* in_sizes, int n_in,
                              void* d_out, int out_size, void* d_ws, size_t ws_size,
                              hipStream_t stream) {
  (void)in_sizes; (void)n_in; (void)out_size; (void)ws_size;
  const float* xyz       = (const float*)d_in[0];
  const float* points    = (const float*)d_in[1];
  const int*   fps_start = (const int*)d_in[2];
  const float* w0 = (const float*)d_in[3];
  const float* g0 = (const float*)d_in[5];
  const float* be0 = (const float*)d_in[6];
  const float* w1 = (const float*)d_in[7];
  const float* g1 = (const float*)d_in[9];
  const float* be1 = (const float*)d_in[10];
  const float* w2 = (const float*)d_in[11];
  const float* g2 = (const float*)d_in[13];
  const float* be2 = (const float*)d_in[14];
  float* out = (float*)d_out;

  char* ws = (char*)d_ws;
  int*       cent   = (int*)(ws + WS_CENT);
  float*     newxyz = (float*)(ws + WS_NEWXYZ);
  float*     stats  = (float*)(ws + WS_STATS);   // 3 layers x 512 floats
  _Float16*  w0p    = (_Float16*)(ws + WS_W0P);
  _Float16*  w1p    = (_Float16*)(ws + WS_W1P);
  _Float16*  w2p    = (_Float16*)(ws + WS_W2P);
  _Float16*  X0     = (_Float16*)(ws + WS_X0);
  _Float16*  Ybuf   = (_Float16*)(ws + WS_Y);
  _Float16*  X1     = (_Float16*)(ws + WS_X1);
  _Float16*  X2     = (_Float16*)(ws + WS_X2);

  const int M = MTOT;

  pack_weights_kernel<<<32, 256, 0, stream>>>(w0, w1, w2, w0p, w1p, w2p, stats);
  fps_kernel<<<BQ, 256, 0, stream>>>(xyz, fps_start, cent, newxyz, out);
  ballquery_gather_kernel<<<BQ * SQ, 256, 0, stream>>>(xyz, points, newxyz, X0);

  // layer 1: 64 x 96(=67 padded)
  gemm_wmma_kernel<<<dim3(M / 128, 4), 256, 0, stream>>>(X0, w0p, Ybuf, stats, M, C0P, 3);
  finalize_stats_kernel<<<1, 128, 0, stream>>>(stats, 1.0f / (float)M);
  norm_relu_pack_kernel<<<(64u * M) / 256, 256, 0, stream>>>(Ybuf, stats, g0, be0, X1, M);

  // layer 2: 64 x 64
  gemm_wmma_kernel<<<dim3(M / 128, 4), 256, 0, stream>>>(X1, w1p, Ybuf, stats + 512, M, 64, 2);
  finalize_stats_kernel<<<1, 128, 0, stream>>>(stats + 512, 1.0f / (float)M);
  norm_relu_pack_kernel<<<(64u * M) / 256, 256, 0, stream>>>(Ybuf, stats + 512, g1, be1, X2, M);

  // layer 3: 128 x 64, fused max over K
  gemm_wmma_kernel<<<dim3(M / 128, 8), 256, 0, stream>>>(X2, w2p, Ybuf, stats + 1024, M, 64, 2);
  finalize_stats_kernel<<<1, 128, 0, stream>>>(stats + 1024, 1.0f / (float)M);
  norm_relu_max_kernel<<<(128u * BQ * SQ) / 256, 256, 0, stream>>>(Ybuf, stats + 1024, g2, be2, out, M);
}
